// GroupedQueryAttention_30734785970221
// MI455X (gfx1250) — compile-verified, matrix operand order checked by reading, not run on hardware
//
#include <hip/hip_runtime.h>
#include <hip/hip_bf16.h>

typedef __bf16 bf16;
typedef __attribute__((ext_vector_type(16))) __bf16 v16bf;
typedef __attribute__((ext_vector_type(8)))  __bf16 v8bf;
typedef __attribute__((ext_vector_type(8)))  float  v8f;

// ---------------------------------------------------------------- helpers

__device__ __forceinline__ bf16 f2bf(float f) {
  union { float f; unsigned u; } x; x.f = f;
  unsigned r = x.u + 0x7FFFu + ((x.u >> 16) & 1u);   // round-to-nearest-even
  union { unsigned short s; bf16 b; } y;
  y.s = (unsigned short)(r >> 16);
  return y.b;
}

__device__ __forceinline__ v8f vzero8() {
  v8f v;
#pragma unroll
  for (int i = 0; i < 8; ++i) v[i] = 0.0f;
  return v;
}

// CDNA5 async copy: each lane moves 16B global -> LDS, tracked by ASYNCcnt.
// GVS mode: addr = SGPR64 base + VGPR 32-bit byte offset.
__device__ __forceinline__ void async_b128(unsigned lds_off, const bf16* base,
                                           unsigned voff) {
  asm volatile("global_load_async_to_lds_b128 %0, %1, %2"
               :: "v"(lds_off), "v"(voff), "s"(base) : "memory");
}
__device__ __forceinline__ void wait_async0() {
  asm volatile("s_wait_asynccnt 0x0" ::: "memory");
}
__device__ __forceinline__ unsigned lds_addr(const bf16* p) {
  return (unsigned)(uintptr_t)p;  // low 32 bits of generic ptr = LDS byte offset
}

// Load one 16x32 bf16 A/B fragment (CDNA5 16-bit operand layout):
// lane L holds row (L&15); K chunk base = (L>>4)*8; elements 0..7 -> K=ks..ks+7,
// elements 8..15 -> K=16+ks..16+ks+7.  Works for LDS and global pointers.
__device__ __forceinline__ v16bf load_frag(const bf16* base, int ld) {
  const int lane = threadIdx.x & 31;
  const bf16* p = base + (size_t)(lane & 15) * ld + ((lane >> 4) << 3);
  v8bf lo = *(const v8bf*)(p);
  v8bf hi = *(const v8bf*)(p + 16);
  union { struct { v8bf lo, hi; } s; v16bf v; } u;
  u.s.lo = lo; u.s.hi = hi;
  return u.v;
}

__device__ __forceinline__ v8f wmma_bf16(v16bf a, v16bf b, v8f c) {
  return __builtin_amdgcn_wmma_f32_16x16x32_bf16(false, a, false, b, (short)0, c,
                                                 false, false);
}

__device__ __forceinline__ float rowred_max(float v) {
#pragma unroll
  for (int m = 1; m <= 8; m <<= 1) v = fmaxf(v, __shfl_xor(v, m, 32));
  return v;
}
__device__ __forceinline__ float rowred_sum(float v) {
#pragma unroll
  for (int m = 1; m <= 8; m <<= 1) v += __shfl_xor(v, m, 32);
  return v;
}

// ---------------------------------------------------------------- pre-pass
// One-shot fp32 -> bf16 conversion (8 elements per thread).
__global__ __launch_bounds__(256) void f32_to_bf16_kernel(
    const float* __restrict__ src, bf16* __restrict__ dst, int n8) {
  const int i = blockIdx.x * 256 + threadIdx.x;
  if (i >= n8) return;
  const float* s = src + (size_t)i * 8;
  float4 f0 = *(const float4*)(s);
  float4 f1 = *(const float4*)(s + 4);
  v8bf c;
  c[0]=f2bf(f0.x); c[1]=f2bf(f0.y); c[2]=f2bf(f0.z); c[3]=f2bf(f0.w);
  c[4]=f2bf(f1.x); c[5]=f2bf(f1.y); c[6]=f2bf(f1.z); c[7]=f2bf(f1.w);
  *(v8bf*)(dst + (size_t)i * 8) = c;
}

// ---------------------------------------------------------------- GEMM
// C[M,N] = A[M,K] @ B[N,K]^T, A/B bf16 in global.  M fixed 4096.
// Block 128x128, 8 waves, wave tile 32x64, K step 32.
// Double-buffered LDS filled by async global->LDS copies.
// OUTMODE 0: out0 bf16 [M,N]
// OUTMODE 1: n<512 -> out0 bf16 [M,512] (K); n>=512 -> out1 bf16 [B,512,T] (V^T)
// OUTMODE 2: out0 f32 [M,N]
template <int OUTMODE, int N, int K>
__global__ __launch_bounds__(256) void gemm_async_wmma(
    const bf16* __restrict__ Ag, const bf16* __restrict__ Bg,
    void* __restrict__ out0, void* __restrict__ out1) {
  constexpr int ALD = 40;  // padded row stride; 80B rows (16B-aligned)
  __shared__ __align__(16) bf16 As[2][128 * ALD];
  __shared__ __align__(16) bf16 Bs[2][128 * ALD];

  const int tid    = threadIdx.x;
  const int lane   = tid & 31;
  const int wave   = tid >> 5;
  const int blockM = blockIdx.y * 128;
  const int blockN = blockIdx.x * 128;
  const int wm     = (wave & 3) * 32;
  const int wn     = (wave >> 2) * 64;

  const int srow = tid >> 1;        // 0..127
  const int sseg = (tid & 1) * 16;  // element offset in 32-wide K slab

  // issue the 4 async 16B copies this thread owns for one k-slab
  auto issue = [&](int buf, int k0) {
    const unsigned aoff = (unsigned)(((blockM + srow) * K + k0 + sseg) * 2);
    const unsigned boff = (unsigned)(((blockN + srow) * K + k0 + sseg) * 2);
    const unsigned la = lds_addr(&As[buf][srow * ALD + sseg]);
    const unsigned lb = lds_addr(&Bs[buf][srow * ALD + sseg]);
    async_b128(la,      Ag, aoff);
    async_b128(la + 16, Ag, aoff + 16);
    async_b128(lb,      Bg, boff);
    async_b128(lb + 16, Bg, boff + 16);
  };

  v8f acc[2][4];
#pragma unroll
  for (int i = 0; i < 2; ++i)
#pragma unroll
    for (int j = 0; j < 4; ++j) acc[i][j] = vzero8();

  issue(0, 0);
  int ib = 0;
  for (int k0 = 0; k0 < K; k0 += 32) {
    wait_async0();        // this wave's async copies landed
    __syncthreads();      // everyone's copies landed; prev readers done
    if (k0 + 32 < K) issue(ib ^ 1, k0 + 32);  // prefetch next slab

    v16bf af[2], bfr[4];
#pragma unroll
    for (int i = 0; i < 2; ++i)
      af[i] = load_frag(&As[ib][(wm + i * 16) * ALD], ALD);
#pragma unroll
    for (int j = 0; j < 4; ++j)
      bfr[j] = load_frag(&Bs[ib][(wn + j * 16) * ALD], ALD);
#pragma unroll
    for (int i = 0; i < 2; ++i)
#pragma unroll
      for (int j = 0; j < 4; ++j)
        acc[i][j] = wmma_bf16(af[i], bfr[j], acc[i][j]);
    ib ^= 1;
  }

  // store (C layout: lane holds col n=lane&15, rows m=(lane>>4)*8+r)
  const int nl = lane & 15;
  const int mh = (lane >> 4) * 8;
#pragma unroll
  for (int i = 0; i < 2; ++i)
#pragma unroll
    for (int j = 0; j < 4; ++j)
#pragma unroll
      for (int r = 0; r < 8; ++r) {
        const int gm  = blockM + wm + i * 16 + mh + r;
        const int gn  = blockN + wn + j * 16 + nl;
        const float v = acc[i][j][r];
        if (OUTMODE == 0) {
          ((bf16*)out0)[(size_t)gm * N + gn] = f2bf(v);
        } else if (OUTMODE == 1) {
          if (gn < 512) {
            ((bf16*)out0)[(size_t)gm * 512 + gn] = f2bf(v);
          } else {  // V stored transposed [B, 512, T]
            const int bb = gm >> 11;
            const int tt = gm & 2047;
            ((bf16*)out1)[((size_t)bb * 512 + (gn - 512)) * 2048 + tt] = f2bf(v);
          }
        } else {
          ((float*)out0)[(size_t)gm * N + gn] = v;
        }
      }
}

// ---------------------------------------------------------------- attention
// One block per (b, head, 128-row query tile). 8 waves; wave owns 16 q-rows.
// Flash-style online softmax over 64-key blocks, causal; K/V^T staged to LDS
// with async global->LDS copies.
__global__ __launch_bounds__(256) void attn_kernel(
    const bf16* __restrict__ Qb,   // [B*T, 2048]  (h*128 + d)
    const bf16* __restrict__ Kb,   // [B*T, 512]   (kvh*128 + d)
    const bf16* __restrict__ Vtb,  // [B, 512, T]  transposed V
    bf16* __restrict__ Ob) {       // [B*T, 2048]
  constexpr float SCALE = 0.08838834764831845f;  // 128^-0.5
  constexpr int KLD = 136, VLD = 72, PLD = 72;   // padded, rows 16B-aligned
  __shared__ __align__(16) bf16 Kt[64 * KLD];     // K block   [64 keys][128 d]
  __shared__ __align__(16) bf16 Vt[128 * VLD];    // V^T block [128 d][64 keys]
  __shared__ __align__(16) bf16 Pt[8 * 16 * PLD]; // per-wave P strip [16 q][64 k]

  const int bz = blockIdx.z, h = blockIdx.y, qt = blockIdx.x;
  const int kvh = h >> 2;
  const int q0  = qt * 128;
  const int tid = threadIdx.x, lane = tid & 31, wave = tid >> 5;
  const int qr0 = q0 + wave * 16;
  const int nl  = lane & 15;
  const int mh  = (lane >> 4) * 8;

  // Q fragments held in registers for all key blocks
  const bf16* Qbase = Qb + (size_t)(bz * 2048 + qr0) * 2048 + h * 128;
  v16bf qf[4];
#pragma unroll
  for (int ks = 0; ks < 4; ++ks) qf[ks] = load_frag(Qbase + ks * 32, 2048);

  v8f oacc[8];
#pragma unroll
  for (int dt = 0; dt < 8; ++dt) oacc[dt] = vzero8();
  float rm[8], rl[8];
#pragma unroll
  for (int r = 0; r < 8; ++r) { rm[r] = -INFINITY; rl[r] = 0.0f; }

  const int nj = (qt + 1) * 2;  // 64-key blocks up to the causal diagonal
  bf16* Pw = &Pt[wave * 16 * PLD];

  const int krow = tid >> 2, kseg = (tid & 3) * 32;  // K stage: 64x128
  const int vrow = tid >> 1, vseg = (tid & 1) * 32;  // V stage: 128x64
  const unsigned lK = lds_addr(&Kt[krow * KLD + kseg]);
  const unsigned lV = lds_addr(&Vt[vrow * VLD + vseg]);

  for (int jb = 0; jb < nj; ++jb) {
    const int kj = jb * 64;
    __syncthreads();  // previous iteration's readers are done with Kt/Vt
    {
      const unsigned g =
          (unsigned)((((bz * 2048 + kj + krow) * 512) + kvh * 128 + kseg) * 2);
#pragma unroll
      for (int c = 0; c < 4; ++c) async_b128(lK + c * 16, Kb, g + c * 16);
    }
    {
      const unsigned g =
          (unsigned)((((bz * 512 + kvh * 128 + vrow) * 2048) + kj + vseg) * 2);
#pragma unroll
      for (int c = 0; c < 4; ++c) async_b128(lV + c * 16, Vtb, g + c * 16);
    }
    wait_async0();
    __syncthreads();

    // ---- S = Q K^T   (16 q-rows x 64 keys per wave)
    v8f sacc[4];
#pragma unroll
    for (int j2 = 0; j2 < 4; ++j2) sacc[j2] = vzero8();
#pragma unroll
    for (int ks = 0; ks < 4; ++ks) {
      v16bf kf[4];
#pragma unroll
      for (int j2 = 0; j2 < 4; ++j2)
        kf[j2] = load_frag(&Kt[(j2 * 16) * KLD + ks * 32], KLD);
#pragma unroll
      for (int j2 = 0; j2 < 4; ++j2)
        sacc[j2] = wmma_bf16(qf[ks], kf[j2], sacc[j2]);
    }

    // ---- scale, causal mask, online softmax
    float pv[4][8];
#pragma unroll
    for (int j2 = 0; j2 < 4; ++j2)
#pragma unroll
      for (int r = 0; r < 8; ++r) {
        float s = sacc[j2][r] * SCALE;
        const int kg = kj + j2 * 16 + nl;
        const int qg = qr0 + mh + r;
        pv[j2][r] = (kg > qg) ? -INFINITY : s;
      }
#pragma unroll
    for (int r = 0; r < 8; ++r) {
      float mx = pv[0][r];
#pragma unroll
      for (int j2 = 1; j2 < 4; ++j2) mx = fmaxf(mx, pv[j2][r]);
      mx = rowred_max(mx);
      const float mnew  = fmaxf(rm[r], mx);
      const float alpha = __expf(rm[r] - mnew);
      rm[r] = mnew;
      float rs = 0.0f;
#pragma unroll
      for (int j2 = 0; j2 < 4; ++j2) {
        const float p = __expf(pv[j2][r] - mnew);
        pv[j2][r] = p;
        rs += p;
      }
      rs = rowred_sum(rs);
      rl[r] = rl[r] * alpha + rs;
#pragma unroll
      for (int dt = 0; dt < 8; ++dt) oacc[dt][r] *= alpha;
    }

    // ---- P: C layout -> A layout via per-wave LDS strip
#pragma unroll
    for (int j2 = 0; j2 < 4; ++j2)
#pragma unroll
      for (int r = 0; r < 8; ++r)
        Pw[(mh + r) * PLD + j2 * 16 + nl] = f2bf(pv[j2][r]);
    asm volatile("s_wait_dscnt 0x0" ::: "memory");  // wave-local visibility

    // ---- O += P V   (keys in two 32-wide slabs)
#pragma unroll
    for (int ks2 = 0; ks2 < 2; ++ks2) {
      v16bf pf = load_frag(Pw + ks2 * 32, PLD);
      v16bf vf[8];
#pragma unroll
      for (int dt = 0; dt < 8; ++dt)
        vf[dt] = load_frag(&Vt[(dt * 16) * VLD + ks2 * 32], VLD);
#pragma unroll
      for (int dt = 0; dt < 8; ++dt)
        oacc[dt] = wmma_bf16(pf, vf[dt], oacc[dt]);
    }
  }

  // ---- normalize + store
  float rinv[8];
#pragma unroll
  for (int r = 0; r < 8; ++r) rinv[r] = 1.0f / rl[r];
#pragma unroll
  for (int dt = 0; dt < 8; ++dt)
#pragma unroll
    for (int r = 0; r < 8; ++r) {
      const int qg = qr0 + mh + r;
      Ob[(size_t)(bz * 2048 + qg) * 2048 + h * 128 + dt * 16 + nl] =
          f2bf(oacc[dt][r] * rinv[r]);
    }
}

// ---------------------------------------------------------------- launch

extern "C" void kernel_launch(void* const* d_in, const int* in_sizes, int n_in,
                              void* d_out, int out_size, void* d_ws, size_t ws_size,
                              hipStream_t stream) {
  (void)in_sizes; (void)n_in; (void)out_size; (void)ws_size;
  const float* x    = (const float*)d_in[0];
  const float* w_q  = (const float*)d_in[1];
  const float* w_kv = (const float*)d_in[2];
  const float* w_o  = (const float*)d_in[3];
  float* out = (float*)d_out;

  char* ws = (char*)d_ws;
  const size_t szQ  = (size_t)4096 * 2048 * sizeof(bf16);    // 16 MB
  const size_t szK  = (size_t)4096 * 512  * sizeof(bf16);    //  4 MB
  const size_t szV  = (size_t)2 * 512 * 2048 * sizeof(bf16); //  4 MB
  const size_t szA  = (size_t)4096 * 2048 * sizeof(bf16);    // 16 MB
  const size_t szX  = (size_t)4096 * 2048 * sizeof(bf16);    // 16 MB
  const size_t szWq = (size_t)2048 * 2048 * sizeof(bf16);    //  8 MB
  const size_t szWk = (size_t)1024 * 2048 * sizeof(bf16);    //  4 MB
  size_t o = 0;
  bf16* Qb   = (bf16*)(ws + o); o += szQ;
  bf16* Kbf  = (bf16*)(ws + o); o += szK;
  bf16* Vtb  = (bf16*)(ws + o); o += szV;
  bf16* Ab   = (bf16*)(ws + o); o += szA;
  bf16* xbf  = (bf16*)(ws + o); o += szX;
  bf16* wqb  = (bf16*)(ws + o); o += szWq;
  bf16* wkvb = (bf16*)(ws + o); o += szWk;
  bf16* wob  = (bf16*)(ws + o);

  // one-shot bf16 conversion of activations + weights
  f32_to_bf16_kernel<<<dim3(1048576 / 256), 256, 0, stream>>>(x,    xbf,  1048576);
  f32_to_bf16_kernel<<<dim3( 524288 / 256), 256, 0, stream>>>(w_q,  wqb,   524288);
  f32_to_bf16_kernel<<<dim3( 262144 / 256), 256, 0, stream>>>(w_kv, wkvb,  262144);
  f32_to_bf16_kernel<<<dim3( 524288 / 256), 256, 0, stream>>>(w_o,  wob,   524288);

  // Q = x @ w_q^T
  gemm_async_wmma<0, 2048, 2048><<<dim3(16, 32), 256, 0, stream>>>(
      xbf, wqb, (void*)Qb, nullptr);
  // KV = x @ w_kv^T ; K -> Kbf, V -> Vtb (transposed)
  gemm_async_wmma<1, 1024, 2048><<<dim3(8, 32), 256, 0, stream>>>(
      xbf, wkvb, (void*)Kbf, (void*)Vtb);
  // causal GQA attention -> Ab
  attn_kernel<<<dim3(16, 16, 2), 256, 0, stream>>>(Qb, Kbf, Vtb, Ab);
  // out = Ab @ w_o^T
  gemm_async_wmma<2, 2048, 2048><<<dim3(16, 32), 256, 0, stream>>>(
      Ab, wob, (void*)out, nullptr);
}
